// Transformer_9251359556062
// MI455X (gfx1250) — matrix operand order repaired and checked by reading, not compiled, not run on hardware
//
#include <hip/hip_runtime.h>

// ---------------------------------------------------------------------------
// Types for CDNA5 WMMA (wave32, v_wmma_f32_16x16x32_f16)
// ---------------------------------------------------------------------------
typedef __attribute__((ext_vector_type(16))) _Float16 v16h;
typedef __attribute__((ext_vector_type(8)))  _Float16 v8h;
typedef __attribute__((ext_vector_type(8)))  float    v8f;

union F16x16 { v16h v; v8h h8[2]; _Float16 h[16]; };

enum : int {
    F_TRB   = 1,    // B element (k,n) = B[n*ldb + k]  (contiguous K loads)
    F_BIAS  = 2,
    F_RELU  = 4,
    F_RESID = 8,
    F_O16   = 16,   // f16 store, row-major
    F_O32   = 32,   // fp32 store, row-major
    F_O16T  = 64,   // f16 store, transposed: C16[col*ldc + row]
    F_BPK   = 128   // B is pre-swizzled into WMMA fragment order (weights)
};

__device__ __forceinline__ void load_a_frag(F16x16& a, const _Float16* __restrict__ pa0, int k)
{
    const _Float16* pa = pa0 + k;
    a.h8[0] = *(const v8h*)(pa);
    a.h8[1] = *(const v8h*)(pa + 8);
}

template<int FLAGS>
__device__ __forceinline__ void load_b_frag(F16x16& b, const _Float16* __restrict__ Bb,
                                            int col0, int t, int k0, int ldb, int KT,
                                            int lane, int r, int ks)
{
    if (FLAGS & F_BPK) {
        // pre-swizzled fragment order: contiguous 32B per lane, lanes contiguous
        const size_t bidx = ((((size_t)((col0 >> 4) + t)) * KT + (k0 >> 5)) * 32 + lane) * 16;
        const _Float16* pb = Bb + bidx;
        b.h8[0] = *(const v8h*)(pb);
        b.h8[1] = *(const v8h*)(pb + 8);
    } else {
        // TRB: (k,n) = B[n*ldb + k] -> contiguous 16-half run per lane
        const _Float16* pb = Bb + (size_t)(col0 + t * 16 + r) * ldb + k0 + ks * 16;
        b.h8[0] = *(const v8h*)(pb);
        b.h8[1] = *(const v8h*)(pb + 8);
    }
}

// ---------------------------------------------------------------------------
// Generic batched WMMA GEMM:  C[M,N] = A[M,K] * B[K,N] (+bias)(+relu)(+resid)
// 4 waves/block, each wave a 16x64 strip -> 64x64 tile/block.
// Software pipeline with two NAMED fragment sets (no dynamic indexing, so
// everything stays in VGPRs): loads for K-step k+32 issue before the 4 wmma
// of step k.  Loop steps by 64 (two 32-phases); all K used are multiples
// of 64, with guards for the tail phases.
// Batch addressing: blockIdx.z -> (bb = z/nh, bh = z%nh) with per-op strides.
// ---------------------------------------------------------------------------
template<int FLAGS>
__global__ __launch_bounds__(128)
void gemm_wmma_k(const _Float16* __restrict__ A, const _Float16* __restrict__ Bm,
                 float* __restrict__ C, _Float16* __restrict__ C16,
                 const float* __restrict__ bias, const float* __restrict__ resid,
                 int M, int N, int K, int lda, int ldb, int ldc,
                 long long oAb, long long oAh, long long oBb, long long oBh,
                 long long oCb, long long oCh, int nh)
{
    const int bz = blockIdx.z;
    const int bb = bz / nh, bh = bz % nh;
    const _Float16* Ab = A  + bb * oAb + bh * oAh;
    const _Float16* Bb = Bm + bb * oBb + bh * oBh;
    const long long oC = bb * oCb + bh * oCh;

    const int lane = threadIdx.x & 31;
    const int wave = threadIdx.x >> 5;
    const int row0 = blockIdx.y * 64 + wave * 16;
    const int col0 = blockIdx.x * 64;
    const int r    = lane & 15;       // row (A) / column (B,C) within tile
    const int ks   = lane >> 4;       // lane-group selector for K split
    const int KT   = K >> 5;          // number of 32-wide K tiles (BPK layout)

    const _Float16* pa0 = Ab + (size_t)(row0 + r) * lda + ks * 16;

    v8f acc[4];
    {
        v8f zz = {};
        #pragma unroll
        for (int t = 0; t < 4; ++t) acc[t] = zz;
    }

    // two named fragment sets -- never dynamically indexed
    F16x16 a0, a1;
    F16x16 b0[4], b1[4];

    // prologue: fragments for k = 0 into set 0
    load_a_frag(a0, pa0, 0);
    #pragma unroll
    for (int t = 0; t < 4; ++t)
        load_b_frag<FLAGS>(b0[t], Bb, col0, t, 0, ldb, KT, lane, r, ks);

    for (int k0 = 0; k0 < K; k0 += 64) {
        const int kA = k0 + 32;   // phase-1 K offset
        const int kB = k0 + 64;   // next iteration's phase-0 K offset

        // phase 0: issue loads for kA into set 1, compute on set 0
        if (kA < K) {
            load_a_frag(a1, pa0, kA);
            #pragma unroll
            for (int t = 0; t < 4; ++t)
                load_b_frag<FLAGS>(b1[t], Bb, col0, t, kA, ldb, KT, lane, r, ks);
        }
        #pragma unroll
        for (int t = 0; t < 4; ++t)
            acc[t] = __builtin_amdgcn_wmma_f32_16x16x32_f16(
                         false, a0.v, false, b0[t].v, (short)0, acc[t], false, false);

        // phase 1: issue loads for kB into set 0, compute on set 1
        if (kB < K) {
            load_a_frag(a0, pa0, kB);
            #pragma unroll
            for (int t = 0; t < 4; ++t)
                load_b_frag<FLAGS>(b0[t], Bb, col0, t, kB, ldb, KT, lane, r, ks);
        }
        if (kA < K) {
            #pragma unroll
            for (int t = 0; t < 4; ++t)
                acc[t] = __builtin_amdgcn_wmma_f32_16x16x32_f16(
                             false, a1.v, false, b1[t].v, (short)0, acc[t], false, false);
        }
    }

    // C/D layout: vgpr j -> row j (lanes 0-15) / row j+8 (lanes 16-31), col = lane&15
    #pragma unroll
    for (int t = 0; t < 4; ++t) {
        const int col = col0 + t * 16 + r;
        if (col >= N) continue;
        #pragma unroll
        for (int j = 0; j < 8; ++j) {
            const int row = row0 + j + ks * 8;
            if (row >= M) continue;
            float v = acc[t][j];
            if (FLAGS & F_BIAS)  v += bias[col];
            if (FLAGS & F_RELU)  v  = fmaxf(v, 0.0f);
            const size_t idx = (size_t)oC + (size_t)row * ldc + col;
            if (FLAGS & F_RESID) v += resid[idx];
            if (FLAGS & F_O32)   C[idx]   = v;
            if (FLAGS & F_O16)   C16[idx] = (_Float16)v;
            if (FLAGS & F_O16T)
                C16[(size_t)oC + (size_t)col * ldc + row] = (_Float16)v;
        }
    }
}

// ---------------------------------------------------------------------------
// Weight conversion + swizzle into WMMA B-fragment order.
// Source: [Lmat, K, N] fp32 row-major.  Dest (per matrix, same elem count):
//   out[(((n/16)*KT + k/32)*32 + lane)*16 + j],  lane = (k%32>=16)*16 + n%16,
//   j = k%16.  In-loop load becomes two contiguous v8h per lane.
// ---------------------------------------------------------------------------
__global__ void packB_k(const float* __restrict__ in, _Float16* __restrict__ out,
                        int K, int N, long long total)
{
    long long o = (long long)blockIdx.x * blockDim.x + threadIdx.x;
    if (o >= total) return;
    const int j    = (int)(o & 15);
    const int lane = (int)((o >> 4) & 31);
    const int KT   = K >> 5;
    const int NT   = N >> 4;
    long long rest = o >> 9;                 // /512
    const int kt   = (int)(rest % KT);
    rest /= KT;
    const int nt   = (int)(rest % NT);
    const long long layer = rest / NT;
    const int k = (kt << 5) + ((lane >> 4) << 4) + j;
    const int n = (nt << 4) + (lane & 15);
    out[o] = (_Float16)in[layer * K * N + (long long)k * N + n];
}

// ---------------------------------------------------------------------------
// LayerNorm: fp32 row in -> f16 row out (feeds WMMA GEMMs)
// ---------------------------------------------------------------------------
__global__ __launch_bounds__(128)
void layernorm_k(const float* __restrict__ x, const float* __restrict__ g,
                 const float* __restrict__ b, _Float16* __restrict__ out, int cols)
{
    __shared__ float s1[128], s2[128];
    const int row = blockIdx.x;
    const int tid = threadIdx.x;
    const float* xr = x + (size_t)row * cols;
    float s = 0.f, q = 0.f;
    for (int i = tid; i < cols; i += 128) { float v = xr[i]; s += v; q += v * v; }
    s1[tid] = s; s2[tid] = q; __syncthreads();
    for (int o = 64; o > 0; o >>= 1) {
        if (tid < o) { s1[tid] += s1[tid + o]; s2[tid] += s2[tid + o]; }
        __syncthreads();
    }
    const float mean = s1[0] / cols;
    const float var  = s2[0] / cols - mean * mean;
    const float rstd = rsqrtf(var + 1e-5f);
    _Float16* orow = out + (size_t)row * cols;
    for (int i = tid; i < cols; i += 128)
        orow[i] = (_Float16)((xr[i] - mean) * rstd * g[i] + b[i]);
}

// ---------------------------------------------------------------------------
// Masked scaled softmax: mode 0 = pad (valid i < len[b]); mode 1 = causal.
// fp32 scores in -> f16 probabilities out.
// ---------------------------------------------------------------------------
__global__ __launch_bounds__(128)
void softmax_k(const float* __restrict__ s, _Float16* __restrict__ p,
               const int* __restrict__ lengths, int Sq, int Sk, int nh,
               int mode, float scale)
{
    __shared__ float red[128];
    const int row = blockIdx.x;           // b*nh*Sq + h*Sq + q
    const int tid = threadIdx.x;
    const int q   = row % Sq;
    const int bi  = row / (nh * Sq);
    const int lim = (mode == 0) ? lengths[bi] : (q + 1);
    const float* sr = s + (size_t)row * Sk;

    float m = -3.0e38f;
    for (int i = tid; i < Sk; i += 128) if (i < lim) m = fmaxf(m, sr[i] * scale);
    red[tid] = m; __syncthreads();
    for (int o = 64; o > 0; o >>= 1) { if (tid < o) red[tid] = fmaxf(red[tid], red[tid + o]); __syncthreads(); }
    m = red[0]; __syncthreads();

    float sum = 0.f;
    for (int i = tid; i < Sk; i += 128) if (i < lim) sum += __expf(sr[i] * scale - m);
    red[tid] = sum; __syncthreads();
    for (int o = 64; o > 0; o >>= 1) { if (tid < o) red[tid] += red[tid + o]; __syncthreads(); }
    const float inv = 1.0f / red[0];

    _Float16* pr = p + (size_t)row * Sk;
    for (int i = tid; i < Sk; i += 128)
        pr[i] = (i < lim) ? (_Float16)(__expf(sr[i] * scale - m) * inv) : (_Float16)0.0f;
}

// ---------------------------------------------------------------------------
// In-place log-softmax over a row of V (final output)
// ---------------------------------------------------------------------------
__global__ __launch_bounds__(256)
void logsoftmax_k(float* __restrict__ x, int V)
{
    __shared__ float red[256];
    const int row = blockIdx.x, tid = threadIdx.x;
    float* xr = x + (size_t)row * V;
    float m = -3.0e38f;
    for (int i = tid; i < V; i += 256) m = fmaxf(m, xr[i]);
    red[tid] = m; __syncthreads();
    for (int o = 128; o > 0; o >>= 1) { if (tid < o) red[tid] = fmaxf(red[tid], red[tid + o]); __syncthreads(); }
    m = red[0]; __syncthreads();
    float s = 0.f;
    for (int i = tid; i < V; i += 256) s += __expf(xr[i] - m);
    red[tid] = s; __syncthreads();
    for (int o = 128; o > 0; o >>= 1) { if (tid < o) red[tid] += red[tid + o]; __syncthreads(); }
    const float lse = m + __logf(red[0]);
    for (int i = tid; i < V; i += 256) xr[i] -= lse;
}

// ---------------------------------------------------------------------------
// Token embedding + sinusoidal positional encoding (computed on the fly)
// ---------------------------------------------------------------------------
__global__ void embed_k(const int* __restrict__ tok, const float* __restrict__ emb,
                        float* __restrict__ out, int S, int HSc, long long n)
{
    long long i = (long long)blockIdx.x * blockDim.x + threadIdx.x;
    if (i >= n) return;
    const int d = (int)(i % HSc);
    const long long ts = i / HSc;        // b*S + s
    const int spos = (int)(ts % S);
    const int t = tok[ts];
    const int half = d >> 1;
    // ang = pos / 10000^(half/HS)
    const float ang = (float)spos * __expf(-((float)half / (float)HSc) * 9.210340371976184f);
    const float pe  = (d & 1) ? __cosf(ang) : __sinf(ang);
    out[i] = emb[(size_t)t * HSc + d] + pe;
}

// ---------------------------------------------------------------------------
// Host orchestration
// ---------------------------------------------------------------------------
#define GEMM(FLAGS_, A_, B_, C_, C16_, BIAS_, RES_, M_, N_, K_, LDA_, LDB_, LDC_, \
             OAB_, OAH_, OBB_, OBH_, OCB_, OCH_, NH_, NZ_)                         \
    gemm_wmma_k<FLAGS_><<<dim3((N_) / 64, (M_) / 64, (NZ_)), 128, 0, stream>>>(    \
        A_, B_, C_, C16_, BIAS_, RES_, M_, N_, K_, LDA_, LDB_, LDC_,               \
        (long long)(OAB_), (long long)(OAH_), (long long)(OBB_),                   \
        (long long)(OBH_), (long long)(OCB_), (long long)(OCH_), NH_)

extern "C" void kernel_launch(void* const* d_in, const int* in_sizes, int n_in,
                              void* d_out, int out_size, void* d_ws, size_t ws_size,
                              hipStream_t stream)
{
    (void)in_sizes; (void)n_in; (void)out_size; (void)ws_size;
    constexpr int Bc = 4, SXc = 512, SYc = 512, HSc = 512, NHc = 8, Lc = 6;
    constexpr int DHc = 64, FFc = 2048, VOUTc = 32000;
    constexpr int Mr = Bc * SXc;  // 2048 rows (same for decoder: Bc*SYc)

    // ---- inputs (setup_inputs dict order) ----
    const int*   x_tok   = (const int*)  d_in[0];
    const int*   lens    = (const int*)  d_in[1];
    const int*   y_tok   = (const int*)  d_in[2];
    const float* emb_enc = (const float*)d_in[3];
    const float* emb_dec = (const float*)d_in[4];
    const float* eWq  = (const float*)d_in[5];
    const float* eWk  = (const float*)d_in[6];
    const float* eWv  = (const float*)d_in[7];
    const float* eWo  = (const float*)d_in[8];
    const float* eln1g= (const float*)d_in[9];
    const float* eln1b= (const float*)d_in[10];
    const float* eln2g= (const float*)d_in[11];
    const float* eln2b= (const float*)d_in[12];
    const float* eW1  = (const float*)d_in[13];
    const float* eb1  = (const float*)d_in[14];
    const float* eW2  = (const float*)d_in[15];
    const float* eb2  = (const float*)d_in[16];
    const float* dsWq = (const float*)d_in[17];
    const float* dsWk = (const float*)d_in[18];
    const float* dsWv = (const float*)d_in[19];
    const float* dsWo = (const float*)d_in[20];
    const float* dcWq = (const float*)d_in[21];
    const float* dcWk = (const float*)d_in[22];
    const float* dcWv = (const float*)d_in[23];
    const float* dcWo = (const float*)d_in[24];
    const float* dln1g= (const float*)d_in[25];
    const float* dln1b= (const float*)d_in[26];
    const float* dln2g= (const float*)d_in[27];
    const float* dln2b= (const float*)d_in[28];
    const float* dln3g= (const float*)d_in[29];
    const float* dln3b= (const float*)d_in[30];
    const float* dW1  = (const float*)d_in[31];
    const float* db1  = (const float*)d_in[32];
    const float* dW2  = (const float*)d_in[33];
    const float* db2  = (const float*)d_in[34];
    const float* glng = (const float*)d_in[35];
    const float* glnb = (const float*)d_in[36];
    const float* gW   = (const float*)d_in[37];
    const float* gb   = (const float*)d_in[38];

    // ---- workspace bump allocator ----
    char* wp = (char*)d_ws;
    auto alloc = [&](size_t bytes) -> void* {
        void* p = (void*)wp;
        wp += (bytes + 255) & ~(size_t)255;
        return p;
    };
    auto h16 = [&](size_t e) -> _Float16* { return (_Float16*)alloc(e * sizeof(_Float16)); };
    auto f32 = [&](size_t e) -> float*    { return (float*)   alloc(e * sizeof(float)); };

    const size_t WHH = (size_t)Lc * HSc * HSc;   // [L,512,512]
    const size_t WHF = (size_t)Lc * HSc * FFc;   // [L,512,2048] / [L,2048,512]

    // f16 weight mirrors (pre-swizzled into WMMA B-fragment order)
    _Float16* eWq6  = h16(WHH); _Float16* eWk6  = h16(WHH);
    _Float16* eWv6  = h16(WHH); _Float16* eWo6  = h16(WHH);
    _Float16* eW16  = h16(WHF); _Float16* eW26  = h16(WHF);
    _Float16* dsWq6 = h16(WHH); _Float16* dsWk6 = h16(WHH);
    _Float16* dsWv6 = h16(WHH); _Float16* dsWo6 = h16(WHH);
    _Float16* dcWq6 = h16(WHH); _Float16* dcWk6 = h16(WHH);
    _Float16* dcWv6 = h16(WHH); _Float16* dcWo6 = h16(WHH);
    _Float16* dW16  = h16(WHF); _Float16* dW26  = h16(WHF);
    _Float16* gW6   = h16((size_t)HSc * VOUTc);

    // activations
    float*    z       = f32((size_t)Mr * HSc);          // encoder stream (fp32)
    float*    h       = f32((size_t)Mr * HSc);          // decoder stream (fp32)
    _Float16* ln16    = h16((size_t)Mr * HSc);
    _Float16* enc16   = h16((size_t)Mr * HSc);          // LN(z, dln2) per layer
    _Float16* q16     = h16((size_t)Mr * HSc);          // [b,s,h,d]
    _Float16* k16     = h16((size_t)Mr * HSc);          // [b,s,h,d]
    _Float16* v16t    = h16((size_t)Mr * HSc);          // [b,h,d,s]  (transposed V)
    _Float16* attn16  = h16((size_t)Mr * HSc);          // [b,s,h,d]
    _Float16* ffn16   = h16((size_t)Mr * FFc);
    float*    scores  = f32((size_t)Bc * NHc * SXc * SXc);
    _Float16* probs   = h16((size_t)Bc * NHc * SXc * SXc);

    // ---- convert + swizzle weights ----
    auto pack = [&](const float* src, _Float16* dst, int K, int N, long long n) {
        packB_k<<<(unsigned)((n + 255) / 256), 256, 0, stream>>>(src, dst, K, N, n);
    };
    pack(eWq, eWq6, HSc, HSc, WHH);  pack(eWk, eWk6, HSc, HSc, WHH);
    pack(eWv, eWv6, HSc, HSc, WHH);  pack(eWo, eWo6, HSc, HSc, WHH);
    pack(eW1, eW16, HSc, FFc, WHF);  pack(eW2, eW26, FFc, HSc, WHF);
    pack(dsWq, dsWq6, HSc, HSc, WHH); pack(dsWk, dsWk6, HSc, HSc, WHH);
    pack(dsWv, dsWv6, HSc, HSc, WHH); pack(dsWo, dsWo6, HSc, HSc, WHH);
    pack(dcWq, dcWq6, HSc, HSc, WHH); pack(dcWk, dcWk6, HSc, HSc, WHH);
    pack(dcWv, dcWv6, HSc, HSc, WHH); pack(dcWo, dcWo6, HSc, HSc, WHH);
    pack(dW1, dW16, HSc, FFc, WHF);  pack(dW2, dW26, FFc, HSc, WHF);
    pack(gW, gW6, HSc, VOUTc, (long long)HSc * VOUTc);

    const long long sQb = (long long)SXc * HSc;   // per-batch stride in q/k/attn (= S*HS)
    const long long sVh = (long long)DHc * SXc;   // per-head stride in v16t
    const long long sSh = (long long)SXc * SXc;   // per-head stride in scores/probs
    const long long sSb = (long long)NHc * sSh;
    const float scale = 0.125f;                   // 1/sqrt(64)

    // =====================  ENCODER  =====================
    embed_k<<<(Mr * HSc + 255) / 256, 256, 0, stream>>>(x_tok, emb_enc, z, SXc, HSc,
                                                        (long long)Mr * HSc);
    for (int i = 0; i < Lc; ++i) {
        const size_t oW = (size_t)i * HSc * HSc;
        const size_t oF = (size_t)i * HSc * FFc;
        layernorm_k<<<Mr, 128, 0, stream>>>(z, eln1g + i * HSc, eln1b + i * HSc, ln16, HSc);
        GEMM(F_BPK | F_O16, ln16, eWq6 + oW, nullptr, q16, nullptr, nullptr,
             Mr, HSc, HSc, HSc, 0, HSc, 0, 0, 0, 0, 0, 0, 1, 1);
        GEMM(F_BPK | F_O16, ln16, eWk6 + oW, nullptr, k16, nullptr, nullptr,
             Mr, HSc, HSc, HSc, 0, HSc, 0, 0, 0, 0, 0, 0, 1, 1);
        // V projection, transposed store -> v16t[b,h,d,s]  (batched over b)
        GEMM(F_BPK | F_O16T, ln16, eWv6 + oW, nullptr, v16t, nullptr, nullptr,
             SXc, HSc, HSc, HSc, 0, SXc, sQb, 0, 0, 0, sQb, 0, 1, Bc);
        // scores = Q @ K^T   (per b,h)
        GEMM(F_TRB | F_O32, q16, k16, scores, nullptr, nullptr, nullptr,
             SXc, SXc, DHc, HSc, HSc, SXc, sQb, DHc, sQb, DHc, sSb, sSh, NHc, Bc * NHc);
        softmax_k<<<Bc * NHc * SXc, 128, 0, stream>>>(scores, probs, lens, SXc, SXc, NHc, 0, scale);
        // context = P @ V   (V consumed transposed -> contiguous K loads)
        GEMM(F_TRB | F_O16, probs, v16t, nullptr, attn16, nullptr, nullptr,
             SXc, DHc, SXc, SXc, SXc, HSc, sSb, sSh, sQb, sVh, sQb, DHc, NHc, Bc * NHc);
        // z = z + context @ Wo
        GEMM(F_BPK | F_RESID | F_O32, attn16, eWo6 + oW, z, nullptr, nullptr, z,
             Mr, HSc, HSc, HSc, 0, HSc, 0, 0, 0, 0, 0, 0, 1, 1);
        // FFN
        layernorm_k<<<Mr, 128, 0, stream>>>(z, eln2g + i * HSc, eln2b + i * HSc, ln16, HSc);
        GEMM(F_BPK | F_O16 | F_BIAS | F_RELU, ln16, eW16 + oF, nullptr, ffn16,
             eb1 + i * FFc, nullptr, Mr, FFc, HSc, HSc, 0, FFc, 0, 0, 0, 0, 0, 0, 1, 1);
        GEMM(F_BPK | F_O32 | F_BIAS | F_RESID, ffn16, eW26 + oF, z, nullptr,
             eb2 + i * HSc, z, Mr, HSc, FFc, FFc, 0, HSc, 0, 0, 0, 0, 0, 0, 1, 1);
    }

    // =====================  DECODER  =====================
    embed_k<<<(Mr * HSc + 255) / 256, 256, 0, stream>>>(y_tok, emb_dec, h, SYc, HSc,
                                                        (long long)Mr * HSc);
    for (int i = 0; i < Lc; ++i) {
        const size_t oW = (size_t)i * HSc * HSc;
        const size_t oF = (size_t)i * HSc * FFc;
        // --- causal self-attention ---
        layernorm_k<<<Mr, 128, 0, stream>>>(h, dln1g + i * HSc, dln1b + i * HSc, ln16, HSc);
        GEMM(F_BPK | F_O16, ln16, dsWq6 + oW, nullptr, q16, nullptr, nullptr,
             Mr, HSc, HSc, HSc, 0, HSc, 0, 0, 0, 0, 0, 0, 1, 1);
        GEMM(F_BPK | F_O16, ln16, dsWk6 + oW, nullptr, k16, nullptr, nullptr,
             Mr, HSc, HSc, HSc, 0, HSc, 0, 0, 0, 0, 0, 0, 1, 1);
        GEMM(F_BPK | F_O16T, ln16, dsWv6 + oW, nullptr, v16t, nullptr, nullptr,
             SYc, HSc, HSc, HSc, 0, SYc, sQb, 0, 0, 0, sQb, 0, 1, Bc);
        GEMM(F_TRB | F_O32, q16, k16, scores, nullptr, nullptr, nullptr,
             SYc, SYc, DHc, HSc, HSc, SYc, sQb, DHc, sQb, DHc, sSb, sSh, NHc, Bc * NHc);
        softmax_k<<<Bc * NHc * SYc, 128, 0, stream>>>(scores, probs, lens, SYc, SYc, NHc, 1, scale);
        GEMM(F_TRB | F_O16, probs, v16t, nullptr, attn16, nullptr, nullptr,
             SYc, DHc, SYc, SYc, SYc, HSc, sSb, sSh, sQb, sVh, sQb, DHc, NHc, Bc * NHc);
        GEMM(F_BPK | F_RESID | F_O32, attn16, dsWo6 + oW, h, nullptr, nullptr, h,
             Mr, HSc, HSc, HSc, 0, HSc, 0, 0, 0, 0, 0, 0, 1, 1);
        // --- cross-attention (kv = LN(z, dln2), q = LN(h, dln2)) ---
        layernorm_k<<<Mr, 128, 0, stream>>>(z, dln2g + i * HSc, dln2b + i * HSc, enc16, HSc);
        layernorm_k<<<Mr, 128, 0, stream>>>(h, dln2g + i * HSc, dln2b + i * HSc, ln16, HSc);
        GEMM(F_BPK | F_O16, ln16,  dcWq6 + oW, nullptr, q16, nullptr, nullptr,
             Mr, HSc, HSc, HSc, 0, HSc, 0, 0, 0, 0, 0, 0, 1, 1);
        GEMM(F_BPK | F_O16, enc16, dcWk6 + oW, nullptr, k16, nullptr, nullptr,
             Mr, HSc, HSc, HSc, 0, HSc, 0, 0, 0, 0, 0, 0, 1, 1);
        GEMM(F_BPK | F_O16T, enc16, dcWv6 + oW, nullptr, v16t, nullptr, nullptr,
             SXc, HSc, HSc, HSc, 0, SXc, sQb, 0, 0, 0, sQb, 0, 1, Bc);
        GEMM(F_TRB | F_O32, q16, k16, scores, nullptr, nullptr, nullptr,
             SYc, SXc, DHc, HSc, HSc, SXc, sQb, DHc, sQb, DHc, sSb, sSh, NHc, Bc * NHc);
        softmax_k<<<Bc * NHc * SYc, 128, 0, stream>>>(scores, probs, lens, SYc, SXc, NHc, 0, scale);
        GEMM(F_TRB | F_O16, probs, v16t, nullptr, attn16, nullptr, nullptr,
             SYc, DHc, SXc, SXc, SXc, HSc, sSb, sSh, sQb, sVh, sQb, DHc, NHc, Bc * NHc);
        GEMM(F_BPK | F_RESID | F_O32, attn16, dcWo6 + oW, h, nullptr, nullptr, h,
             Mr, HSc, HSc, HSc, 0, HSc, 0, 0, 0, 0, 0, 0, 1, 1);
        // --- FFN ---
        layernorm_k<<<Mr, 128, 0, stream>>>(h, dln3g + i * HSc, dln3b + i * HSc, ln16, HSc);
        GEMM(F_BPK | F_O16 | F_BIAS | F_RELU, ln16, dW16 + oF, nullptr, ffn16,
             db1 + i * FFc, nullptr, Mr, FFc, HSc, HSc, 0, FFc, 0, 0, 0, 0, 0, 0, 1, 1);
        GEMM(F_BPK | F_O32 | F_BIAS | F_RESID, ffn16, dW26 + oF, h, nullptr,
             db2 + i * HSc, h, Mr, HSc, FFc, FFc, 0, HSc, 0, 0, 0, 0, 0, 0, 1, 1);
    }

    // =====================  OUTPUT HEAD  =====================
    layernorm_k<<<Mr, 128, 0, stream>>>(h, glng, glnb, ln16, HSc);
    GEMM(F_BPK | F_O32 | F_BIAS, ln16, gW6, (float*)d_out, nullptr, gb, nullptr,
         Mr, VOUTc, HSc, HSc, 0, VOUTc, 0, 0, 0, 0, 0, 0, 1, 1);
    logsoftmax_k<<<Mr, 256, 0, stream>>>((float*)d_out, VOUTc);
}
